// NeighbourAssignment_79680233275437
// MI455X (gfx1250) — compile-verified, matrix operand order checked by reading, not run on hardware
//
#include <hip/hip_runtime.h>
#include <stdint.h>

#define NN   16384   // target nodes
#define NS   16384   // source nodes
#define KNB  32      // neighbors per node
#define CSZ  256     // source channels
#define CTZ  256     // target channels
#define COZ  64      // out channels
#define MMZ  64      // number of mixtures (== CO)

typedef __attribute__((ext_vector_type(16))) __bf16 v16bf;
typedef __attribute__((ext_vector_type(8)))  float  v8f;
typedef __attribute__((ext_vector_type(8)))  unsigned int v8u;

__device__ __forceinline__ unsigned short f32_to_bf16(float f) {
  unsigned int u = __float_as_uint(f);
  u = (u + 0x7FFFu + ((u >> 16) & 1u)) >> 16;   // round-to-nearest-even
  return (unsigned short)u;
}

__device__ __forceinline__ unsigned int pk_bf16(float lo, float hi) {
  unsigned int ulo = __float_as_uint(lo);
  unsigned int uhi = __float_as_uint(hi);
  ulo = (ulo + 0x7FFFu + ((ulo >> 16) & 1u)) >> 16;
  uhi = (uhi + 0x7FFFu + ((uhi >> 16) & 1u)) & 0xFFFF0000u;
  return (ulo & 0xFFFFu) | uhi;
}

// XOR lane swizzle within 32 lanes: offset = {xor_mask[14:10], or=0, and=0x1F}
__device__ __forceinline__ float swz_xor8(float v) {
  return __int_as_float(__builtin_amdgcn_ds_swizzle(__float_as_int(v), 0x201F));
}
__device__ __forceinline__ float swz_xor4(float v) {
  return __int_as_float(__builtin_amdgcn_ds_swizzle(__float_as_int(v), 0x101F));
}
__device__ __forceinline__ float swz_xor2(float v) {
  return __int_as_float(__builtin_amdgcn_ds_swizzle(__float_as_int(v), 0x081F));
}
__device__ __forceinline__ float swz_xor1(float v) {
  return __int_as_float(__builtin_amdgcn_ds_swizzle(__float_as_int(v), 0x041F));
}

// ---------------------------------------------------------------------------
// Prep 0: source f32 [NS][CS] -> bf16 (packed u32) once; each gathered row is
// reused ~K times, so converting up front removes all cvt VALU from the hot loop
// and halves the L2 gather traffic.
// ---------------------------------------------------------------------------
__global__ void __launch_bounds__(256)
prep_src(const float* __restrict__ src, uint2* __restrict__ dst) {
  int tid = blockIdx.x * blockDim.x + threadIdx.x;   // NS*CS/4 threads
  float4 v = ((const float4*)src)[tid];
  uint2 o;
  o.x = pk_bf16(v.x, v.y);
  o.y = pk_bf16(v.z, v.w);
  dst[tid] = o;
}

// ---------------------------------------------------------------------------
// Prep 1: W_mean = mean_m(W_lin) transposed -> bf16 [64][256]
//         W_s transposed -> bf16 [64][256];  b_mean = mean_m(b_lin)
// ---------------------------------------------------------------------------
__global__ void __launch_bounds__(256)
prep_weights(const float* __restrict__ W_lin, const float* __restrict__ b_lin,
             const float* __restrict__ W_s,
             unsigned short* __restrict__ Ws_t, unsigned short* __restrict__ Wm_t,
             float* __restrict__ b_mean) {
  int tid = blockIdx.x * blockDim.x + threadIdx.x;   // 64*256 threads
  int o = tid >> 8;        // 0..63
  int c = tid & 255;       // 0..255
  float acc = 0.f;
  for (int m = 0; m < MMZ; ++m)
    acc += W_lin[(size_t)m * (CSZ * COZ) + (size_t)c * COZ + o];
  acc *= (1.0f / (float)MMZ);
  Wm_t[o * CSZ + c] = f32_to_bf16(acc);
  Ws_t[o * CSZ + c] = f32_to_bf16(W_s[c * MMZ + o]);
  if (c == 0) {
    float b = 0.f;
    for (int m = 0; m < MMZ; ++m) b += b_lin[m * COZ + o];
    b_mean[o] = b * (1.0f / (float)MMZ);
  }
}

// ---------------------------------------------------------------------------
// Prep 2: t_logit[n][m] = target[n] . W_t[:,m] + b_t[m] + b_s[m]   (f32)
// ---------------------------------------------------------------------------
__global__ void __launch_bounds__(256)
prep_tlogit(const float* __restrict__ target, const float* __restrict__ W_t,
            const float* __restrict__ b_t, const float* __restrict__ b_s,
            float* __restrict__ t_logit) {
  int tid = blockIdx.x * blockDim.x + threadIdx.x;   // N*64 threads
  int n = tid >> 6;
  int m = tid & 63;
  const float* tg = target + (size_t)n * CTZ;
  float acc = b_t[m] + b_s[m];
  for (int c = 0; c < CTZ; ++c)
    acc = fmaf(tg[c], W_t[c * MMZ + m], acc);
  t_logit[tid] = acc;
}

// ---------------------------------------------------------------------------
// Fused: bf16 gather + dual bf16 WMMA GEMM + softmax(M) + elementwise product
// 256 threads = 8 waves; each wave owns a 16-row tile of the (N*K) rows.
// ---------------------------------------------------------------------------
#define LDS_ROW 264   // 256 + 8 bf16 pad: row stride 528B -> 4-bank skew, conflict-free b128

__global__ void __launch_bounds__(256)
fused_assign(const unsigned short* __restrict__ src_bf,   // bf16 [NS][CS]
             const int*   __restrict__ neighbors,
             const unsigned short* __restrict__ Ws_t,     // bf16 [64][256]
             const unsigned short* __restrict__ Wm_t,     // bf16 [64][256]
             const float* __restrict__ b_mean,            // [64]
             const float* __restrict__ t_logit,           // [N][64] (biases folded in)
             float* __restrict__ out)                     // [N*K][64]
{
  extern __shared__ unsigned short smem[];
  unsigned short* sWs = smem;
  unsigned short* sWm = smem + 64 * LDS_ROW;

  // Stage both transposed weight matrices into padded LDS (once per block).
  {
    const uint32_t* g0 = (const uint32_t*)Ws_t;
    const uint32_t* g1 = (const uint32_t*)Wm_t;
    uint32_t* d0 = (uint32_t*)sWs;
    uint32_t* d1 = (uint32_t*)sWm;
    for (int idx = threadIdx.x; idx < 64 * 128; idx += 256) {
      int row = idx >> 7, col = idx & 127;
      d0[row * (LDS_ROW / 2) + col] = g0[idx];
      d1[row * (LDS_ROW / 2) + col] = g1[idx];
    }
  }
  __syncthreads();

  const int lane = threadIdx.x & 31;
  const int wave = threadIdx.x >> 5;
  const int half = lane >> 4;     // K-half selector for A/B operand layout
  const int r    = lane & 15;     // row (A) / column (B,D) index within tile

  const int tile_base = blockIdx.x * 128 + wave * 16;  // 16 | 32 -> n constant per tile
  const int n = tile_base >> 5;
  const int src_row = neighbors[tile_base + r];
  const unsigned short* srcp = src_bf + (size_t)src_row * CSZ;

  // Accumulators: 4 column-tiles of logits (s) and 4 of projection (p).
  v8f s_acc[4], p_acc[4];
  const float* tl = t_logit + (size_t)n * MMZ;
#pragma unroll
  for (int tc = 0; tc < 4; ++tc) {
    float sv = tl[tc * 16 + r];       // bias + target logit (constant down the column)
    float pv = b_mean[tc * 16 + r];
#pragma unroll
    for (int i = 0; i < 8; ++i) { s_acc[tc][i] = sv; p_acc[tc][i] = pv; }
  }

  // K loop: CS=256 in 8 steps of 32 channels.
#pragma unroll 2
  for (int ks = 0; ks < 8; ++ks) {
    const int kbase = ks * 32;

    // A operand: per-lane bf16 gather (L2-resident), already in operand layout.
    // lanes 0-15: K = kbase+[0,8) U kbase+16+[0,8); lanes 16-31: +8 offset.
    const uint4* ap = (const uint4*)(srcp + kbase + half * 8);
    uint4 a0 = ap[0], a1 = ap[2];          // +16 elements = +32 B
    v8u apk;
    apk[0] = a0.x; apk[1] = a0.y; apk[2] = a0.z; apk[3] = a0.w;
    apk[4] = a1.x; apk[5] = a1.y; apk[6] = a1.z; apk[7] = a1.w;
    v16bf A = __builtin_bit_cast(v16bf, apk);

#pragma unroll
    for (int tc = 0; tc < 4; ++tc) {
      const int brow = tc * 16 + r;   // B column (transposed weight row)

      const uint4* bs = (const uint4*)(sWs + brow * LDS_ROW + kbase + half * 8);
      uint4 b0 = bs[0], b1 = bs[2];
      v8u bpk;
      bpk[0] = b0.x; bpk[1] = b0.y; bpk[2] = b0.z; bpk[3] = b0.w;
      bpk[4] = b1.x; bpk[5] = b1.y; bpk[6] = b1.z; bpk[7] = b1.w;
      v16bf Bs = __builtin_bit_cast(v16bf, bpk);
      s_acc[tc] = __builtin_amdgcn_wmma_f32_16x16x32_bf16(
          false, A, false, Bs, (short)0, s_acc[tc], false, false);

      const uint4* bm = (const uint4*)(sWm + brow * LDS_ROW + kbase + half * 8);
      uint4 c0 = bm[0], c1 = bm[2];
      v8u cpk;
      cpk[0] = c0.x; cpk[1] = c0.y; cpk[2] = c0.z; cpk[3] = c0.w;
      cpk[4] = c1.x; cpk[5] = c1.y; cpk[6] = c1.z; cpk[7] = c1.w;
      v16bf Bm = __builtin_bit_cast(v16bf, cpk);
      p_acc[tc] = __builtin_amdgcn_wmma_f32_16x16x32_bf16(
          false, A, false, Bm, (short)0, p_acc[tc], false, false);
    }
  }

  // Softmax over M=64 per output row, in the WMMA D layout:
  // row i's 64 columns live in 16 lanes of one half-wave across the 4 tc tiles.
  // XOR swizzles with masks 8/4/2/1 stay within each 16-lane column group.
#pragma unroll
  for (int i = 0; i < 8; ++i) {
    float mx = fmaxf(fmaxf(s_acc[0][i], s_acc[1][i]),
                     fmaxf(s_acc[2][i], s_acc[3][i]));
    mx = fmaxf(mx, swz_xor8(mx));
    mx = fmaxf(mx, swz_xor4(mx));
    mx = fmaxf(mx, swz_xor2(mx));
    mx = fmaxf(mx, swz_xor1(mx));
    float e0 = __expf(s_acc[0][i] - mx);
    float e1 = __expf(s_acc[1][i] - mx);
    float e2 = __expf(s_acc[2][i] - mx);
    float e3 = __expf(s_acc[3][i] - mx);
    float sum = (e0 + e1) + (e2 + e3);
    sum += swz_xor8(sum);
    sum += swz_xor4(sum);
    sum += swz_xor2(sum);
    sum += swz_xor1(sum);
    float rs = 1.0f / sum;
    s_acc[0][i] = e0 * rs; s_acc[1][i] = e1 * rs;
    s_acc[2][i] = e2 * rs; s_acc[3][i] = e3 * rs;
  }

  // out = assignment * proj (layouts identical). Lanes 0-15 store rows 0-7,
  // lanes 16-31 rows 8-15; each (i,tc) store is 64B contiguous across lanes.
  const int out_row0 = tile_base + half * 8;
#pragma unroll
  for (int i = 0; i < 8; ++i) {
    float* orow = out + (size_t)(out_row0 + i) * COZ;
#pragma unroll
    for (int tc = 0; tc < 4; ++tc)
      orow[tc * 16 + r] = s_acc[tc][i] * p_acc[tc][i];
  }
}

// ---------------------------------------------------------------------------
extern "C" void kernel_launch(void* const* d_in, const int* in_sizes, int n_in,
                              void* d_out, int out_size, void* d_ws, size_t ws_size,
                              hipStream_t stream) {
  const float* source    = (const float*)d_in[0];
  const float* target    = (const float*)d_in[1];
  const int*   neighbors = (const int*)  d_in[2];
  const float* W_lin     = (const float*)d_in[3];
  const float* b_lin     = (const float*)d_in[4];
  const float* W_s       = (const float*)d_in[5];
  const float* b_s       = (const float*)d_in[6];
  const float* W_t       = (const float*)d_in[7];
  const float* b_t       = (const float*)d_in[8];
  float* out = (float*)d_out;

  char* ws = (char*)d_ws;
  unsigned short* Ws_t   = (unsigned short*)(ws);                 // 32768 B
  unsigned short* Wm_t   = (unsigned short*)(ws + 32768);         // 32768 B
  float*          b_mean = (float*)(ws + 65536);                  // 256 B
  float*          tlog   = (float*)(ws + 65792);                  // 4 MB
  unsigned short* src_bf = (unsigned short*)(ws + 65792 + (size_t)NN * MMZ * 4); // 8 MB

  prep_src<<<(NS * CSZ / 4) / 256, 256, 0, stream>>>(source, (uint2*)src_bf);
  prep_weights<<<(MMZ * CSZ) / 256, 256, 0, stream>>>(W_lin, b_lin, W_s,
                                                      Ws_t, Wm_t, b_mean);
  prep_tlogit<<<(NN * MMZ) / 256, 256, 0, stream>>>(target, W_t, b_t, b_s, tlog);

  size_t smem = 2u * 64u * LDS_ROW * sizeof(unsigned short); // 67584 B
  fused_assign<<<(NN * KNB) / 128, 256, smem, stream>>>(src_bf, neighbors,
                                                        Ws_t, Wm_t, b_mean,
                                                        tlog, out);
}
